// CLSTM_61314953117850
// MI455X (gfx1250) — matrix operand order repaired and checked by reading, not compiled, not run on hardware
//
#include <hip/hip_runtime.h>
#include <hip/hip_bf16.h>
#include <math.h>

typedef __attribute__((ext_vector_type(16))) _Float16 v16h;
typedef __attribute__((ext_vector_type(8)))  _Float16 v8h;
typedef __attribute__((ext_vector_type(8)))  float    v8f;

#define B_SZ 1024
#define T_SZ 3072
#define S_SZ 1536

__device__ __forceinline__ float sigf(float x) { return 1.f / (1.f + __expf(-x)); }

// 8 WMMAs for a 32x64 register tile (2 A-frags x 4 B-frags)
__device__ __forceinline__ void mma8(const v16h a[2], const v16h b[4], v8f acc[2][4]) {
#pragma unroll
  for (int i = 0; i < 2; ++i)
#pragma unroll
    for (int j = 0; j < 4; ++j)
      acc[i][j] = __builtin_amdgcn_wmma_f32_16x16x32_f16(
          false, a[i], false, b[j], (short)0, acc[i][j], false, false);
}

// Pack 16 f32 (as 4x float4) into a v16h fragment (v_cvt_pk_f16_f32 pairs)
__device__ __forceinline__ v16h pack16f(float4 p0, float4 p1, float4 p2, float4 p3) {
  v16h r;
  r[0]=(_Float16)p0.x;  r[1]=(_Float16)p0.y;  r[2]=(_Float16)p0.z;  r[3]=(_Float16)p0.w;
  r[4]=(_Float16)p1.x;  r[5]=(_Float16)p1.y;  r[6]=(_Float16)p1.z;  r[7]=(_Float16)p1.w;
  r[8]=(_Float16)p2.x;  r[9]=(_Float16)p2.y;  r[10]=(_Float16)p2.z; r[11]=(_Float16)p2.w;
  r[12]=(_Float16)p3.x; r[13]=(_Float16)p3.y; r[14]=(_Float16)p3.z; r[15]=(_Float16)p3.w;
  return r;
}

// ---------------------------------------------------------------------------
// C[M,N] = act(A[M,K] @ W[N,K]^T + bias[N]); one wave32 per 32x64 C tile.
// Vector path (K%32==0): per-lane 128-bit loads; scalar guarded path otherwise.
// STORE_T stores C transposed (C[n*M+m]) to feed the time-major LSTM scan.
// Fragment layouts used here (contiguous K order, same for A and B):
//   A 16x32 f16 : lane = M row, half h -> K=16g+h
//   B 32x16 f16 : lane = N col, half h -> K=16g+h
//   C f32       : reg r -> M=r+8g, N=lane&15
// ---------------------------------------------------------------------------
template<bool LEAKY, bool STORE_T>
__global__ __launch_bounds__(256) void gemm_wmma_k(
    const float* __restrict__ A, const float* __restrict__ W,
    const float* __restrict__ bias, float* __restrict__ C,
    int M, int N, int K)
{
  const int wave = blockIdx.x * (blockDim.x >> 5) + (threadIdx.x >> 5);
  const int ntn  = N >> 6;                  // 64-wide N tiles (all N % 64 == 0)
  const int tm = wave / ntn, tn = wave % ntn;
  if (tm * 32 >= M) return;
  const int lane = threadIdx.x & 31;
  const int g = lane >> 4, l15 = lane & 15;

  v8f acc[2][4] = {};

  if ((K & 31) == 0) {                      // fast path: 128-bit loads
    for (int k0 = 0; k0 < K; k0 += 32) {
      v16h a[2], b[4];
#pragma unroll
      for (int i = 0; i < 2; ++i) {
        const float* ap = A + (size_t)(tm * 32 + i * 16 + l15) * K + k0 + 16 * g;
        a[i] = pack16f(*(const float4*)(ap),      *(const float4*)(ap + 4),
                       *(const float4*)(ap + 8),  *(const float4*)(ap + 12));
      }
#pragma unroll
      for (int j = 0; j < 4; ++j) {
        const float* wp = W + (size_t)(tn * 64 + j * 16 + l15) * K + k0 + 16 * g;
        b[j] = pack16f(*(const float4*)(wp),     *(const float4*)(wp + 4),
                       *(const float4*)(wp + 8), *(const float4*)(wp + 12));
      }
      mma8(a, b, acc);
    }
  } else {                                  // ragged K (stem layer 0, K=5)
    for (int k0 = 0; k0 < K; k0 += 32) {
      v16h a[2], b[4];
#pragma unroll
      for (int i = 0; i < 2; ++i) {
        const int row = tm * 32 + i * 16 + l15;
#pragma unroll
        for (int h = 0; h < 16; ++h) {
          int ka = k0 + 16 * g + h;
          a[i][h] = (_Float16)((ka < K) ? A[(size_t)row * K + ka] : 0.f);
        }
      }
#pragma unroll
      for (int j = 0; j < 4; ++j) {
        const int col = tn * 64 + j * 16 + l15;
#pragma unroll
        for (int h = 0; h < 16; ++h) {
          int kb = k0 + 16 * g + h;
          b[j][h] = (_Float16)((kb < K) ? W[(size_t)col * K + kb] : 0.f);
        }
      }
      mma8(a, b, acc);
    }
  }

#pragma unroll
  for (int i = 0; i < 2; ++i)
#pragma unroll
    for (int j = 0; j < 4; ++j)
#pragma unroll
      for (int r = 0; r < 8; ++r) {
        int m = tm * 32 + i * 16 + r + 8 * g;
        int n = tn * 64 + j * 16 + l15;
        float v = acc[i][j][r] + bias[n];
        if (LEAKY) v = (v > 0.f) ? v : 0.2f * v;
        if (STORE_T) C[(size_t)n * M + m] = v;
        else         C[(size_t)m * N + n] = v;
      }
}

// ---------------------------------------------------------------------------
// Final einsum 'kbt,kst->kbs': per head, C[b,s] = sum_t ho[b,t]*outW[s,t]+outb[s]
// ho is f16 [4][B][T] (K-contiguous): A fragment = two b128 loads per lane.
// ---------------------------------------------------------------------------
__global__ __launch_bounds__(256) void gemm_out_k(
    const _Float16* __restrict__ ho,  // [4][B][T]
    const float* __restrict__ outW,   // [4][S][T]
    const float* __restrict__ outb,   // [4][S]
    float* __restrict__ out,          // [4][B][S]
    int M, int N, int K)              // M=B, N=S, K=T
{
  const int head = blockIdx.y;
  const _Float16* Ah = ho   + (size_t)head * M * K;
  const float*    Wh = outW + (size_t)head * N * K;
  const float*    bh = outb + (size_t)head * N;
  float*          Ch = out  + (size_t)head * M * N;

  const int wave = blockIdx.x * (blockDim.x >> 5) + (threadIdx.x >> 5);
  const int ntn  = N >> 6;
  const int tm = wave / ntn, tn = wave % ntn;
  if (tm * 32 >= M) return;
  const int lane = threadIdx.x & 31;
  const int g = lane >> 4, l15 = lane & 15;

  v8f acc[2][4] = {};
  for (int k0 = 0; k0 < K; k0 += 32) {
    v16h a[2], b[4];
#pragma unroll
    for (int i = 0; i < 2; ++i) {
      const _Float16* ap = Ah + (size_t)(tm * 32 + i * 16 + l15) * K + k0 + 16 * g;
      v8h lo = *(const v8h*)(ap);
      v8h hi = *(const v8h*)(ap + 8);
#pragma unroll
      for (int h = 0; h < 8; ++h) { a[i][h] = lo[h]; a[i][8 + h] = hi[h]; }
    }
#pragma unroll
    for (int j = 0; j < 4; ++j) {
      const float* wp = Wh + (size_t)(tn * 64 + j * 16 + l15) * K + k0 + 16 * g;
      b[j] = pack16f(*(const float4*)(wp),     *(const float4*)(wp + 4),
                     *(const float4*)(wp + 8), *(const float4*)(wp + 12));
    }
    mma8(a, b, acc);
  }
#pragma unroll
  for (int i = 0; i < 2; ++i)
#pragma unroll
    for (int j = 0; j < 4; ++j)
#pragma unroll
      for (int r = 0; r < 8; ++r) {
        int m = tm * 32 + i * 16 + r + 8 * g;
        int n = tn * 64 + j * 16 + l15;
        Ch[(size_t)m * N + n] = acc[i][j][r] + bh[n];
      }
}

// ---------------------------------------------------------------------------
// Bidirectional micro-LSTM (input=1, hidden=4): one thread per (dir, batch).
// seqT is time-major [T,B] so each step's load is lane-coalesced.
// Gate order i,f,g,o (torch). Writes y as [T,B,8] (fwd 0..3, bwd 4..7).
// ---------------------------------------------------------------------------
__global__ __launch_bounds__(256) void lstm0_k(
    const float* __restrict__ seqT,
    const float* __restrict__ WihF, const float* __restrict__ WhhF,
    const float* __restrict__ bihF, const float* __restrict__ bhhF,
    const float* __restrict__ WihB, const float* __restrict__ WhhB,
    const float* __restrict__ bihB, const float* __restrict__ bhhB,
    const float* __restrict__ h0m, const float* __restrict__ c0m,
    float* __restrict__ y)
{
  const int dir = blockIdx.y;
  __shared__ float sWih[16], sWhh[64], sB[16];
  const float* Wih = dir ? WihB : WihF;
  const float* Whh = dir ? WhhB : WhhF;
  const float* bih = dir ? bihB : bihF;
  const float* bhh = dir ? bhhB : bhhF;
  if (threadIdx.x < 16) {
    sWih[threadIdx.x] = Wih[threadIdx.x];
    sB[threadIdx.x]   = bih[threadIdx.x] + bhh[threadIdx.x];
  }
  if (threadIdx.x < 64) sWhh[threadIdx.x] = Whh[threadIdx.x];
  __syncthreads();

  const int b = blockIdx.x * blockDim.x + threadIdx.x;
  float h[4], c[4];
#pragma unroll
  for (int j = 0; j < 4; ++j) {
    h[j] = h0m[(size_t)dir * B_SZ * 4 + b * 4 + j];
    c[j] = c0m[(size_t)dir * B_SZ * 4 + b * 4 + j];
  }
  for (int s = 0; s < T_SZ; ++s) {
    const int t = dir ? (T_SZ - 1 - s) : s;
    const float x = seqT[(size_t)t * B_SZ + b];
    if (s + 32 < T_SZ) {                    // global_prefetch_b8 for a future step
      const int tp = dir ? (T_SZ - 1 - (s + 32)) : (s + 32);
      __builtin_prefetch(&seqT[(size_t)tp * B_SZ + b], 0, 0);
    }
    float nh[4];
#pragma unroll
    for (int j = 0; j < 4; ++j) {
      float pi = sWih[j]      * x + sB[j];
      float pf = sWih[4 + j]  * x + sB[4 + j];
      float pg = sWih[8 + j]  * x + sB[8 + j];
      float po = sWih[12 + j] * x + sB[12 + j];
#pragma unroll
      for (int u = 0; u < 4; ++u) {
        pi += sWhh[j * 4 + u]        * h[u];
        pf += sWhh[(4 + j) * 4 + u]  * h[u];
        pg += sWhh[(8 + j) * 4 + u]  * h[u];
        po += sWhh[(12 + j) * 4 + u] * h[u];
      }
      c[j]  = sigf(pf) * c[j] + sigf(pi) * tanhf(pg);
      nh[j] = sigf(po) * tanhf(c[j]);
    }
#pragma unroll
    for (int j = 0; j < 4; ++j) h[j] = nh[j];
    float4 hv = make_float4(nh[0], nh[1], nh[2], nh[3]);
    *(float4*)(y + ((size_t)t * B_SZ + b) * 8 + dir * 4) = hv;
  }
}

// ---------------------------------------------------------------------------
// 4 head LSTMs (input=8, hidden=1): one thread per (head, batch).
// Buffers 16 steps in registers, then writes 2x b128 to ho f16 [4][B][T].
// ---------------------------------------------------------------------------
__global__ __launch_bounds__(256) void heads_k(
    const float* __restrict__ y,      // [T,B,8]
    const float* __restrict__ Wih,    // [4,4,8]
    const float* __restrict__ Whh,    // [4,4]
    const float* __restrict__ bih, const float* __restrict__ bhh, // [4,4]
    const float* __restrict__ h0h, const float* __restrict__ c0h, // [4,B]
    _Float16* __restrict__ ho)        // [4,B,T]
{
  const int k = blockIdx.y;
  __shared__ float sW[32], sU[4], sB[4];
  if (threadIdx.x < 32) sW[threadIdx.x] = Wih[k * 32 + threadIdx.x];
  if (threadIdx.x < 4) {
    sU[threadIdx.x] = Whh[k * 4 + threadIdx.x];
    sB[threadIdx.x] = bih[k * 4 + threadIdx.x] + bhh[k * 4 + threadIdx.x];
  }
  __syncthreads();

  const int b = blockIdx.x * blockDim.x + threadIdx.x;
  float h = h0h[(size_t)k * B_SZ + b];
  float c = c0h[(size_t)k * B_SZ + b];
  _Float16* hok = ho + ((size_t)k * B_SZ + b) * T_SZ;
  _Float16 hb[16];
  for (int t = 0; t < T_SZ; ++t) {
    const float4* yp = (const float4*)(y + ((size_t)t * B_SZ + b) * 8);
    float4 y0 = yp[0], y1 = yp[1];
    float yv[8] = {y0.x, y0.y, y0.z, y0.w, y1.x, y1.y, y1.z, y1.w};
    float pre[4];
#pragma unroll
    for (int gg = 0; gg < 4; ++gg) {
      float p = sB[gg] + sU[gg] * h;
#pragma unroll
      for (int u = 0; u < 8; ++u) p += sW[gg * 8 + u] * yv[u];
      pre[gg] = p;
    }
    c = sigf(pre[1]) * c + sigf(pre[0]) * tanhf(pre[2]);
    h = sigf(pre[3]) * tanhf(c);
    hb[t & 15] = (_Float16)h;
    if ((t & 15) == 15) {                   // flush 16 steps = 32B
      v8h lo, hi;
#pragma unroll
      for (int u = 0; u < 8; ++u) { lo[u] = hb[u]; hi[u] = hb[8 + u]; }
      *(v8h*)(hok + (t - 15))     = lo;
      *(v8h*)(hok + (t - 15) + 8) = hi;
    }
  }
}

// ---------------------------------------------------------------------------
extern "C" void kernel_launch(void* const* d_in, const int* in_sizes, int n_in,
                              void* d_out, int out_size, void* d_ws, size_t ws_size,
                              hipStream_t stream) {
  (void)in_sizes; (void)n_in; (void)out_size; (void)ws_size;
  const float* x = (const float*)d_in[0];
  const float* Ws[7]; const float* bs[7];
  for (int i = 0; i < 7; ++i) {
    Ws[i] = (const float*)d_in[1 + 2 * i];
    bs[i] = (const float*)d_in[2 + 2 * i];
  }
  const float* WihF = (const float*)d_in[15];
  const float* WhhF = (const float*)d_in[16];
  const float* bihF = (const float*)d_in[17];
  const float* bhhF = (const float*)d_in[18];
  const float* WihB = (const float*)d_in[19];
  const float* WhhB = (const float*)d_in[20];
  const float* bihB = (const float*)d_in[21];
  const float* bhhB = (const float*)d_in[22];
  const float* hWih = (const float*)d_in[23];
  const float* hWhh = (const float*)d_in[24];
  const float* hbih = (const float*)d_in[25];
  const float* hbhh = (const float*)d_in[26];
  const float* outW = (const float*)d_in[27];
  const float* outb = (const float*)d_in[28];
  const float* h0m  = (const float*)d_in[29];
  const float* c0m  = (const float*)d_in[30];
  const float* h0h  = (const float*)d_in[31];
  const float* c0h  = (const float*)d_in[32];

  // Workspace layout
  char* wsb = (char*)d_ws;
  float*    bufA = (float*)wsb;                                     // 1024*3072 f32
  float*    bufB = (float*)(wsb + (size_t)B_SZ * T_SZ * 4);         // 1024*1536 f32
  float*    ybuf = (float*)(wsb + (size_t)B_SZ * T_SZ * 4
                                + (size_t)B_SZ * S_SZ * 4);         // [T,B,8] f32
  _Float16* hob  = (_Float16*)(wsb + (size_t)B_SZ * T_SZ * 4
                                   + (size_t)B_SZ * S_SZ * 4
                                   + (size_t)T_SZ * B_SZ * 8 * 4);  // [4,B,T] f16

  // Stem: 6 leaky-ReLU GEMMs + final GEMM stored transposed [T,B]
  const int kd[7] = {5, 64, 128, 256, 512, 1024, 1536};
  const int nd[7] = {64, 128, 256, 512, 1024, 1536, 3072};
  const float* cur = x;
  for (int i = 0; i < 7; ++i) {
    float* outp = (i & 1) ? bufB : bufA;
    int tiles  = (B_SZ / 32) * (nd[i] / 64);
    int blocks = (tiles * 32 + 255) / 256;
    if (i < 6)
      gemm_wmma_k<true, false><<<blocks, 256, 0, stream>>>(cur, Ws[i], bs[i], outp,
                                                           B_SZ, nd[i], kd[i]);
    else
      gemm_wmma_k<false, true><<<blocks, 256, 0, stream>>>(cur, Ws[i], bs[i], outp,
                                                           B_SZ, nd[i], kd[i]);
    cur = outp;
  }
  // bufA now holds seq^T [T,B]

  lstm0_k<<<dim3(B_SZ / 256, 2), 256, 0, stream>>>(bufA,
      WihF, WhhF, bihF, bhhF, WihB, WhhB, bihB, bhhB, h0m, c0m, ybuf);

  heads_k<<<dim3(B_SZ / 256, 4), 256, 0, stream>>>(ybuf,
      hWih, hWhh, hbih, hbhh, h0h, c0h, hob);

  {
    int tiles  = (B_SZ / 32) * (S_SZ / 64);   // 768 waves per head
    int blocks = (tiles * 32 + 255) / 256;    // 96
    gemm_out_k<<<dim3(blocks, 4), 256, 0, stream>>>(hob, outW, outb,
                                                    (float*)d_out, B_SZ, S_SZ, T_SZ);
  }
}
